// FusionBlock_71751723647636
// MI455X (gfx1250) — matrix operand order repaired and checked by reading, not compiled, not run on hardware
//
#include <hip/hip_runtime.h>
#include <hip/hip_bf16.h>

typedef __bf16 v8bf  __attribute__((ext_vector_type(8)));
typedef __bf16 v16bf __attribute__((ext_vector_type(16)));
typedef float  v8f   __attribute__((ext_vector_type(8)));

#define DMODEL 256
#define LTXT   256
#define NHEAD  4
#define HD     64
#define LDS_LD 264   // padded row stride in bf16 elements (33*16B, bank-rotating)

#define WAIT_LDS() asm volatile("s_wait_dscnt 0" ::: "memory")

__device__ __forceinline__ unsigned short f2bf(float x){
  unsigned int u = __builtin_bit_cast(unsigned int, x);
  u += 0x7fffu + ((u >> 16) & 1u);            // round-to-nearest-even
  return (unsigned short)(u >> 16);
}

__device__ __forceinline__ v8f zero8(){
  v8f z;
#pragma unroll
  for(int i=0;i<8;i++) z[i]=0.0f;
  return z;
}

// A-operand (16x32 bf16, MxK): lane m=lane&15, h=lane>>4.
// elements 0..7  -> K = 8h + e       (contiguous run)
// elements 8..15 -> K = 16 + 8h + (e-8)
__device__ __forceinline__ v16bf load_a_frag(const unsigned short* rowk, int h){
  v8bf lo = *(const v8bf*)(rowk + 8*h);
  v8bf hi = *(const v8bf*)(rowk + 16 + 8*h);
  v16bf r;
#pragma unroll
  for(int i=0;i<8;i++){ r[i]=lo[i]; r[i+8]=hi[i]; }
  return r;
}

// B-operand (32x16 bf16, KxN): lane n=lane&15, h=lane>>4.
// elements 0..7 -> K = 8h + e, elements 8..15 -> K = 16 + 8h + (e-8)
__device__ __forceinline__ v16bf load_b_frag(const unsigned short* rowk, int h){
  v8bf lo = *(const v8bf*)(rowk + 8*h);
  v8bf hi = *(const v8bf*)(rowk + 16 + 8*h);
  v16bf r;
#pragma unroll
  for(int i=0;i<8;i++){ r[i]=lo[i]; r[i+8]=hi[i]; }
  return r;
}

__device__ __forceinline__ v8f wmma_bf16(v16bf a, v16bf b, v8f c){
  return __builtin_amdgcn_wmma_f32_16x16x32_bf16(false, a, false, b, (short)0, c, false, false);
}

// ---------------- small setup kernels ----------------

__global__ void k_init(float* out, int* counts, int nout, int B){
  int i = blockIdx.x*blockDim.x + threadIdx.x;
  if (i < nout) out[i] = 0.0f;
  if (i < B)    counts[i] = 0;
}

__global__ void k_count(const int* sb, int* counts, int n){
  int i = blockIdx.x*blockDim.x + threadIdx.x;
  if (i < n) atomicAdd(&counts[sb[i]], 1);
}

__global__ void k_meta(const int* counts, int B, int* offsets, float* invL,
                       int2* tiles, int* tilecount){
  if (blockIdx.x == 0 && threadIdx.x == 0){
    int off = 0, t = 0;
    for (int b = 0; b < B; b++){
      offsets[b] = off;
      int L = counts[b];
      invL[b] = 1.0f / fmaxf((float)L, 1.0f);
      for (int s = 0; s < L; s += 16) tiles[t++] = make_int2(b, off + s);
      off += L;
    }
    offsets[B] = off;
    *tilecount = t;
  }
}

__global__ void k_cvt(const float* in, unsigned short* out, int n){
  for (int i = blockIdx.x*blockDim.x + threadIdx.x; i < n; i += gridDim.x*blockDim.x)
    out[i] = f2bf(in[i]);
}

// ---------------- K/V projection: one wave per (batch, 16-token tile) ----------------

__global__ __launch_bounds__(64)
void k_kv(const float* __restrict__ text, const unsigned short* __restrict__ wk,
          const unsigned short* __restrict__ wv, const float* __restrict__ in_b,
          unsigned short* __restrict__ Kout, unsigned short* __restrict__ Vt, int B){
  __shared__ unsigned short buf[2][16*LDS_LD];
  int wave = threadIdx.x >> 5, lane = threadIdx.x & 31;
  int gw = blockIdx.x*2 + wave;
  if (gw >= B*16) return;
  int b = gw >> 4, t0 = (gw & 15) * 16;
  unsigned short* A = buf[wave];
  const float* src = text + ((size_t)b*LTXT + t0)*DMODEL;
  for (int idx = lane; idx < 16*DMODEL; idx += 32){
    int r = idx >> 8, c = idx & 255;
    A[r*LDS_LD + c] = f2bf(src[r*DMODEL + c]);
  }
  WAIT_LDS();
  int l15 = lane & 15, h = lane >> 4;
  const float* bk = in_b + DMODEL;
  const float* bv = in_b + 2*DMODEL;
#pragma unroll 1
  for (int nt = 0; nt < 16; nt++){
    int ng = nt*16 + l15;
    v8f accK = zero8(), accV = zero8();
    const unsigned short* wkr = wk + (size_t)ng*DMODEL;
    const unsigned short* wvr = wv + (size_t)ng*DMODEL;
#pragma unroll
    for (int kc = 0; kc < 8; kc++){
      v16bf a = load_a_frag(A + l15*LDS_LD + kc*32, h);
      accK = wmma_bf16(a, load_b_frag(wkr + kc*32, h), accK);
      accV = wmma_bf16(a, load_b_frag(wvr + kc*32, h), accV);
    }
    float biK = bk[ng], biV = bv[ng];
#pragma unroll
    for (int r = 0; r < 8; r++){
      int t = t0 + r + 8*h;
      Kout[((size_t)b*LTXT + t)*DMODEL + ng] = f2bf(accK[r] + biK);   // token-major
      Vt [((size_t)b*DMODEL + ng)*LTXT + t] = f2bf(accV[r] + biV);    // d-major
    }
  }
}

// ---------------- fused attention mega-kernel: one wave per 16-query tile ----------------

__global__ __launch_bounds__(64)
void k_attn(const float* __restrict__ sfeat, const int* __restrict__ offsets,
            const int2* __restrict__ tiles, const int* __restrict__ tilecount,
            const float* __restrict__ invL,
            const unsigned short* __restrict__ wq, const unsigned short* __restrict__ wo,
            const unsigned short* __restrict__ wf,
            const float* __restrict__ in_b, const float* __restrict__ out_b,
            const float* __restrict__ ln_w, const float* __restrict__ ln_b,
            const float* __restrict__ fc_b,
            const unsigned short* __restrict__ Kmat, const unsigned short* __restrict__ Vt,
            float* __restrict__ out){
  __shared__ unsigned short bufA_s[2][16*LDS_LD];
  __shared__ unsigned short bufB_s[2][16*LDS_LD];
  int wave = threadIdx.x >> 5, lane = threadIdx.x & 31;
  int ti = blockIdx.x*2 + wave;
  if (ti >= *tilecount) return;
  int2 tile = tiles[ti];
  int b = tile.x, row0 = tile.y, segend = offsets[b+1];
  unsigned short* A  = bufA_s[wave];
  unsigned short* Bb = bufB_s[wave];
  int l15 = lane & 15, h = lane >> 4;

  // stage 0: struct rows -> LDS bf16 (pad invalid rows with 0)
  for (int idx = lane; idx < 16*DMODEL; idx += 32){
    int r = idx >> 8, c = idx & 255;
    float v = (row0 + r < segend) ? sfeat[(size_t)(row0 + r)*DMODEL + c] : 0.0f;
    A[r*LDS_LD + c] = f2bf(v);
  }
  WAIT_LDS();

  // stage 1: Q = rows @ Wq^T + bq  -> Bb (bf16); unroll 2 => dual WMMA chains
#pragma unroll 2
  for (int nt = 0; nt < 16; nt++){
    int ng = nt*16 + l15;
    v8f acc = zero8();
    const unsigned short* wr = wq + (size_t)ng*DMODEL;
#pragma unroll
    for (int kc = 0; kc < 8; kc++)
      acc = wmma_bf16(load_a_frag(A + l15*LDS_LD + kc*32, h), load_b_frag(wr + kc*32, h), acc);
    float bi = in_b[ng];
#pragma unroll
    for (int r = 0; r < 8; r++) Bb[(r + 8*h)*LDS_LD + ng] = f2bf(acc[r] + bi);
  }
  WAIT_LDS();

  const unsigned short* Kb  = Kmat + (size_t)b*LTXT*DMODEL;
  const unsigned short* Vb  = Vt   + (size_t)b*DMODEL*LTXT;

  // stage 2: per-head attention; ctx overwrites Q's head columns in Bb
#pragma unroll 1
  for (int hh = 0; hh < NHEAD; hh++){
    float s[128];
#pragma unroll
    for (int nt = 0; nt < 16; nt++){
      v8f acc = zero8();
      const unsigned short* kr = Kb + (size_t)(nt*16 + l15)*DMODEL + hh*HD;
#pragma unroll
      for (int kc = 0; kc < 2; kc++)
        acc = wmma_bf16(load_a_frag(Bb + l15*LDS_LD + hh*HD + kc*32, h),
                        load_b_frag(kr + kc*32, h), acc);
#pragma unroll
      for (int r = 0; r < 8; r++) s[nt*8 + r] = acc[r] * 0.125f;   // 1/sqrt(64)
    }
    // softmax over 256 keys per row (rows 0-7 on lanes 0-15, rows 8-15 on lanes 16-31)
    float rmax[8], rsum[8];
#pragma unroll
    for (int r = 0; r < 8; r++) rmax[r] = -1e30f;
#pragma unroll
    for (int nt = 0; nt < 16; nt++)
#pragma unroll
      for (int r = 0; r < 8; r++) rmax[r] = fmaxf(rmax[r], s[nt*8 + r]);
#pragma unroll
    for (int r = 0; r < 8; r++){
#pragma unroll
      for (int m = 1; m < 16; m <<= 1) rmax[r] = fmaxf(rmax[r], __shfl_xor(rmax[r], m, 32));
      rsum[r] = 0.0f;
    }
#pragma unroll
    for (int nt = 0; nt < 16; nt++)
#pragma unroll
      for (int r = 0; r < 8; r++){
        float e = __expf(s[nt*8 + r] - rmax[r]);
        s[nt*8 + r] = e; rsum[r] += e;
      }
#pragma unroll
    for (int r = 0; r < 8; r++){
#pragma unroll
      for (int m = 1; m < 16; m <<= 1) rsum[r] += __shfl_xor(rsum[r], m, 32);
      rsum[r] = 1.0f / rsum[r];
    }
    // P -> A (bf16, row-major over 256 keys)
#pragma unroll
    for (int nt = 0; nt < 16; nt++){
      int col = nt*16 + l15;
#pragma unroll
      for (int r = 0; r < 8; r++) A[(r + 8*h)*LDS_LD + col] = f2bf(s[nt*8 + r] * rsum[r]);
    }
    WAIT_LDS();
    // ctx_head = P @ V_head  (B-operand rows are d-major V rows: contiguous over tokens)
#pragma unroll 2
    for (int dt = 0; dt < 4; dt++){
      int ng = hh*HD + dt*16 + l15;
      v8f acc = zero8();
      const unsigned short* vr = Vb + (size_t)ng*LTXT;
#pragma unroll
      for (int kc = 0; kc < 8; kc++)
        acc = wmma_bf16(load_a_frag(A + l15*LDS_LD + kc*32, h), load_b_frag(vr + kc*32, h), acc);
#pragma unroll
      for (int r = 0; r < 8; r++) Bb[(r + 8*h)*LDS_LD + ng] = f2bf(acc[r]);
    }
    WAIT_LDS();
  }

  // stage 3: fused = ctx @ Wout^T + b  (keep f32 in registers for LN)
  float f[128];
#pragma unroll
  for (int nt = 0; nt < 16; nt++){
    int ng = nt*16 + l15;
    v8f acc = zero8();
    const unsigned short* wr = wo + (size_t)ng*DMODEL;
#pragma unroll
    for (int kc = 0; kc < 8; kc++)
      acc = wmma_bf16(load_a_frag(Bb + l15*LDS_LD + kc*32, h), load_b_frag(wr + kc*32, h), acc);
    float bi = out_b[ng];
#pragma unroll
    for (int r = 0; r < 8; r++) f[nt*8 + r] = acc[r] + bi;
  }

  // stage 4: LayerNorm over d=256 per row
  float mu[8], sq[8];
#pragma unroll
  for (int r = 0; r < 8; r++){ mu[r] = 0.0f; sq[r] = 0.0f; }
#pragma unroll
  for (int nt = 0; nt < 16; nt++)
#pragma unroll
    for (int r = 0; r < 8; r++){ float v = f[nt*8 + r]; mu[r] += v; sq[r] += v*v; }
#pragma unroll
  for (int r = 0; r < 8; r++){
#pragma unroll
    for (int m = 1; m < 16; m <<= 1){
      mu[r] += __shfl_xor(mu[r], m, 32);
      sq[r] += __shfl_xor(sq[r], m, 32);
    }
    mu[r] *= (1.0f/DMODEL);
    float var = sq[r]*(1.0f/DMODEL) - mu[r]*mu[r];
    sq[r] = rsqrtf(var + 1e-5f);           // reuse as rstd
  }
#pragma unroll
  for (int nt = 0; nt < 16; nt++){
    int ng = nt*16 + l15;
    float gw = ln_w[ng], gb = ln_b[ng];
#pragma unroll
    for (int r = 0; r < 8; r++)
      A[(r + 8*h)*LDS_LD + ng] = f2bf((f[nt*8 + r] - mu[r])*sq[r]*gw + gb);
  }
  WAIT_LDS();

  // stage 5: fc + masked per-batch mean accumulation; unroll 2 => dual WMMA chains
  float invl = invL[b];
#pragma unroll 2
  for (int nt = 0; nt < 16; nt++){
    int ng = nt*16 + l15;
    v8f acc = zero8();
    const unsigned short* wr = wf + (size_t)ng*DMODEL;
#pragma unroll
    for (int kc = 0; kc < 8; kc++)
      acc = wmma_bf16(load_a_frag(A + l15*LDS_LD + kc*32, h), load_b_frag(wr + kc*32, h), acc);
    float bi = fc_b[ng];
    float sm = 0.0f;
#pragma unroll
    for (int r = 0; r < 8; r++)
      if (row0 + r + 8*h < segend) sm += acc[r] + bi;
    sm += __shfl_xor(sm, 16, 32);          // combine both row-halves (same d)
    if (h == 0) atomicAdd(&out[(size_t)b*DMODEL + ng], sm * invl);
  }
}

// ---------------- host launch ----------------

extern "C" void kernel_launch(void* const* d_in, const int* in_sizes, int n_in,
                              void* d_out, int out_size, void* d_ws, size_t ws_size,
                              hipStream_t stream) {
  const float* struct_feat = (const float*)d_in[0];
  const float* text_feat   = (const float*)d_in[1];
  const int*   struct_batch= (const int*)  d_in[2];
  const float* in_proj_w   = (const float*)d_in[3];
  const float* in_proj_b   = (const float*)d_in[4];
  const float* out_proj_w  = (const float*)d_in[5];
  const float* out_proj_b  = (const float*)d_in[6];
  const float* ln_w        = (const float*)d_in[7];
  const float* ln_b        = (const float*)d_in[8];
  const float* fc_w        = (const float*)d_in[9];
  const float* fc_b        = (const float*)d_in[10];
  float* out = (float*)d_out;

  int N = in_sizes[0] / DMODEL;
  int B = in_sizes[1] / (LTXT * DMODEL);
  int maxtiles = (N + 15)/16 + B;

  char* ws = (char*)d_ws;
  size_t o = 0;
  unsigned short* Kmat = (unsigned short*)(ws + o); o += (size_t)B*LTXT*DMODEL*2;
  unsigned short* Vt   = (unsigned short*)(ws + o); o += (size_t)B*DMODEL*LTXT*2;
  unsigned short* wqkv = (unsigned short*)(ws + o); o += (size_t)3*DMODEL*DMODEL*2;
  unsigned short* wob  = (unsigned short*)(ws + o); o += (size_t)DMODEL*DMODEL*2;
  unsigned short* wfb  = (unsigned short*)(ws + o); o += (size_t)DMODEL*DMODEL*2;
  int*   counts   = (int*)  (ws + o); o += 256;
  int*   offsets  = (int*)  (ws + o); o += 512;
  float* invL     = (float*)(ws + o); o += 256;
  int*   tilecount= (int*)  (ws + o); o += 256;
  int2*  tiles    = (int2*) (ws + o);

  k_init <<<(B*DMODEL + 255)/256, 256, 0, stream>>>(out, counts, B*DMODEL, B);
  k_count<<<(N + 255)/256,        256, 0, stream>>>(struct_batch, counts, N);
  k_meta <<<1, 1, 0, stream>>>(counts, B, offsets, invL, tiles, tilecount);
  k_cvt  <<<192, 256, 0, stream>>>(in_proj_w,  wqkv, 3*DMODEL*DMODEL);
  k_cvt  <<<128, 256, 0, stream>>>(out_proj_w, wob,  DMODEL*DMODEL);
  k_cvt  <<<128, 256, 0, stream>>>(fc_w,       wfb,  DMODEL*DMODEL);
  k_kv   <<<(B*16 + 1)/2, 64, 0, stream>>>(text_feat, wqkv + DMODEL*DMODEL,
                                           wqkv + 2*DMODEL*DMODEL, in_proj_b, Kmat, Vt, B);
  k_attn <<<(maxtiles + 1)/2, 64, 0, stream>>>(struct_feat, offsets, tiles, tilecount, invL,
                                               wqkv, wob, wfb, in_proj_b, out_proj_b,
                                               ln_w, ln_b, fc_b, Kmat, Vt, out);
}
